// MultiHeadAttention_74749610820014
// MI455X (gfx1250) — matrix operand order repaired and checked by reading, not compiled, not run on hardware
//
#include <hip/hip_runtime.h>

// ---------------------------------------------------------------------------
// MI455X (gfx1250) fused multi-head attention:
//   qkv GEMM (bf16 WMMA, async-LDS double-buffered) -> RoPE + V-transpose
//   -> causal flash attention (bf16 WMMA) -> output projection GEMM (fp32 out)
// wave32; v_wmma_f32_16x16x32_bf16 + global_load_async_to_lds_b128.
// ---------------------------------------------------------------------------

typedef __bf16 bf16_t;
typedef __attribute__((ext_vector_type(16))) __bf16 v16bf;
typedef __attribute__((ext_vector_type(8)))  __bf16 v8bf;
typedef __attribute__((ext_vector_type(8)))  float  v8f;

#define DIM       2048
#define N_HEADS   16
#define HEAD_DIM  128
#define BATCH     2
#define SEQ       4096
#define ROWS      (BATCH * SEQ)        // 8192
#define QKV_COLS  (3 * DIM)            // 6144
#define KDEPTH    64                   // K-slab staged per async round

__device__ __forceinline__ v8f wmma_bf16(v16bf a, v16bf b, v8f c) {
  // D = A(16x32) * B(32x16) + C(16x16), fp32 accum
  return __builtin_amdgcn_wmma_f32_16x16x32_bf16(
      false, a, false, b, (short)0, c, false, false);
}

// CDNA5 async copy: 16 bytes per lane, global -> LDS, tracked by ASYNCcnt.
__device__ __forceinline__ void async_b128(void* lds_ptr, const bf16_t* gp) {
  uint32_t l = (uint32_t)(uintptr_t)lds_ptr;     // low 32 bits = LDS offset
  uint64_t g = (uint64_t)(uintptr_t)gp;
  asm volatile("global_load_async_to_lds_b128 %0, %1, off"
               :: "v"(l), "v"(g) : "memory");
}
__device__ __forceinline__ void wait_async_all() {
  asm volatile("s_wait_asynccnt 0x0" ::: "memory");
}

// ---------------------------------------------------------------------------
// WMMA fragment loaders (generic pointers: work for global and LDS sources).
// A (16x32): lane<16 -> M=lane,    K in {0..7, 16..23}
//            lane>=16 -> M=lane-16, K in {8..15, 24..31}
// ---------------------------------------------------------------------------
__device__ __forceinline__ v16bf load_frag_a(const bf16_t* base, int ld,
                                             int m0, int k0) {
  int lane = threadIdx.x & 31;
  int half = lane >> 4;
  const bf16_t* p = base + (size_t)(m0 + (lane & 15)) * ld + k0 + half * 8;
  v8bf lo = *(const v8bf*)p;
  v8bf hi = *(const v8bf*)(p + 16);
  v16bf r;
#pragma unroll
  for (int i = 0; i < 8; ++i) { r[i] = lo[i]; r[8 + i] = hi[i]; }
  return r;
}

// B (32x16): column n = row (n0+n) of `base` along K.
// lane<16 -> col=lane, K in {0..7, 16..23} ; lane>=16 -> col=lane-16, K in {8..15, 24..31}
__device__ __forceinline__ v16bf load_frag_b(const bf16_t* base, int ld,
                                             int n0, int k0) {
  int lane = threadIdx.x & 31;
  int half = lane >> 4;
  const bf16_t* p = base + (size_t)(n0 + (lane & 15)) * ld + k0 + half * 8;
  v8bf lo = *(const v8bf*)p;
  v8bf hi = *(const v8bf*)(p + 16);
  v16bf r;
#pragma unroll
  for (int i = 0; i < 8; ++i) { r[i] = lo[i]; r[8 + i] = hi[i]; }
  return r;
}

// ---------------------------------------------------------------------------
// fp32 -> bf16 conversion
// ---------------------------------------------------------------------------
__global__ __launch_bounds__(256) void f32_to_bf16_k(const float* __restrict__ src,
                                                     bf16_t* __restrict__ dst,
                                                     int n) {
  int stride = gridDim.x * blockDim.x;
  for (int i = blockIdx.x * blockDim.x + threadIdx.x; i < n; i += stride)
    dst[i] = (bf16_t)src[i];
}

// ---------------------------------------------------------------------------
// GEMM: C[M,N] = A[M,K] * W[N,K]^T
// block = 256 threads = 8 waves; wave -> 16x64 tile; block -> 128x64.
// A/B K-slabs (128x64, 64x64 bf16) staged in LDS via async copies,
// double-buffered so copies of slab i+1 overlap WMMAs on slab i.
// ---------------------------------------------------------------------------
__device__ __forceinline__ void issue_slab(const bf16_t* A, const bf16_t* W,
                                           bf16_t (*sAb)[KDEPTH],
                                           bf16_t (*sBb)[KDEPTH],
                                           int tid, int mblk, int nblk,
                                           int K, int k0) {
  int row = tid >> 3;            // 0..31
  int seg = (tid & 7) * 8;       // 16B segment within 64-elem row
#pragma unroll
  for (int i = 0; i < 4; ++i) {  // A: 128 rows
    int r = row + i * 32;
    async_b128(&sAb[r][seg], A + (size_t)(mblk + r) * K + k0 + seg);
  }
#pragma unroll
  for (int i = 0; i < 2; ++i) {  // B: 64 rows
    int r = row + i * 32;
    async_b128(&sBb[r][seg], W + (size_t)(nblk + r) * K + k0 + seg);
  }
}

template <bool OUT_BF16>
__global__ __launch_bounds__(256) void gemm_wmma_k(const bf16_t* __restrict__ A,
                                                   const bf16_t* __restrict__ W,
                                                   void* __restrict__ C,
                                                   int M, int N, int K) {
  __shared__ bf16_t sA[2][128][KDEPTH];   // 2 x 16KB
  __shared__ bf16_t sB[2][64][KDEPTH];    // 2 x  8KB

  int tid = threadIdx.x;
  int wave = tid >> 5;
  int lane = tid & 31;
  int half = lane >> 4;
  int ncol = lane & 15;
  int mblk = blockIdx.x * 128;
  int nblk = blockIdx.y * 64;
  int m0 = wave * 16;                     // wave's rows within the block tile

  int nslabs = K / KDEPTH;

  // prologue: stage slab 0
  issue_slab(A, W, sA[0], sB[0], tid, mblk, nblk, K, 0);
  wait_async_all();
  __syncthreads();

  v8f acc[4] = {};
  for (int it = 0; it < nslabs; ++it) {
    int buf = it & 1;
    if (it + 1 < nslabs)
      issue_slab(A, W, sA[buf ^ 1], sB[buf ^ 1], tid, mblk, nblk, K,
                 (it + 1) * KDEPTH);

#pragma unroll
    for (int kk = 0; kk < KDEPTH; kk += 32) {
      v16bf a = load_frag_a(&sA[buf][0][0], KDEPTH, m0, kk);
#pragma unroll
      for (int j = 0; j < 4; ++j) {
        v16bf b = load_frag_b(&sB[buf][0][0], KDEPTH, j * 16, kk);
        acc[j] = wmma_bf16(a, b, acc[j]);
      }
    }

    wait_async_all();     // our async writes into buf^1 are done
    __syncthreads();      // everyone finished reading buf, writes visible
  }

  // C layout: lane<16 -> (M=r, N=ncol); lane>=16 -> (M=r+8, N=ncol)
#pragma unroll
  for (int j = 0; j < 4; ++j) {
#pragma unroll
    for (int r = 0; r < 8; ++r) {
      int m = mblk + m0 + r + half * 8;
      int c = nblk + j * 16 + ncol;
      if (OUT_BF16)
        ((bf16_t*)C)[(size_t)m * N + c] = (bf16_t)acc[j][r];
      else
        ((float*)C)[(size_t)m * N + c] = acc[j][r];
    }
  }
}

// ---------------------------------------------------------------------------
// RoPE (in-place on Q,K columns of qkv) + V transpose into vt[b,h,d,t]
// ---------------------------------------------------------------------------
__global__ __launch_bounds__(256) void rope_reorg_k(bf16_t* __restrict__ qkv,
                                                    bf16_t* __restrict__ vt) {
  int row = blockIdx.x;            // b*SEQ + t
  int b = row >> 12;               // SEQ = 4096
  int t = row & (SEQ - 1);
  float tf = (float)t;
  bf16_t* qrow = qkv + (size_t)row * QKV_COLS;

  for (int p = threadIdx.x; p < 1024; p += blockDim.x) {
    int h = p >> 6;
    int i = p & 63;
    int c0 = h * HEAD_DIM + 2 * i;
    int d0 = 2 * i, d1 = 2 * i + 1;
    float a0 = tf * __powf(10000.0f, -(float)(d0 & 63) * (1.0f / 64.0f));
    float a1 = tf * __powf(10000.0f, -(float)(d1 & 63) * (1.0f / 64.0f));
    float c0f = __cosf(a0), s0f = __sinf(a0);
    float c1f = __cosf(a1), s1f = __sinf(a1);
    float x0 = (float)qrow[c0],      x1 = (float)qrow[c0 + 1];
    qrow[c0]     = (bf16_t)(x0 * c0f - x1 * s0f);
    qrow[c0 + 1] = (bf16_t)(x1 * c1f + x0 * s1f);
    float y0 = (float)qrow[DIM + c0], y1 = (float)qrow[DIM + c0 + 1];
    qrow[DIM + c0]     = (bf16_t)(y0 * c0f - y1 * s0f);
    qrow[DIM + c0 + 1] = (bf16_t)(y1 * c1f + y0 * s1f);
  }
  for (int e = threadIdx.x; e < DIM; e += blockDim.x) {
    int h = e >> 7;
    int d = e & (HEAD_DIM - 1);
    vt[((size_t)(b * N_HEADS + h) * HEAD_DIM + d) * SEQ + t] =
        qrow[2 * DIM + e];
  }
}

// ---------------------------------------------------------------------------
// Flash attention, causal. block = 8 waves; wave -> 16 query rows.
// grid = (SEQ/128, B*N_HEADS).
// ---------------------------------------------------------------------------
__global__ __launch_bounds__(256) void attention_k(const bf16_t* __restrict__ qkv,
                                                   const bf16_t* __restrict__ vt,
                                                   bf16_t* __restrict__ obf) {
  __shared__ bf16_t sP[8][16][32];   // per-wave P staging (C-layout -> A-layout)

  int wave = threadIdx.x >> 5;
  int lane = threadIdx.x & 31;
  int half = lane >> 4;
  int ncol = lane & 15;

  int bh = blockIdx.y;
  int b = bh >> 4;                   // N_HEADS = 16
  int h = bh & 15;
  int q0 = blockIdx.x * 128 + wave * 16;

  const bf16_t* Qb = qkv + (size_t)b * SEQ * QKV_COLS + h * HEAD_DIM;
  const bf16_t* Kb = Qb + DIM;
  const bf16_t* Vh = vt + (size_t)bh * HEAD_DIM * SEQ;   // [128, SEQ]

  v16bf qf[4];
#pragma unroll
  for (int d = 0; d < 4; ++d) qf[d] = load_frag_a(Qb, QKV_COLS, q0, d * 32);

  v8f o[8] = {};
  float rmax[8], rsum[8];
#pragma unroll
  for (int r = 0; r < 8; ++r) { rmax[r] = -1e30f; rsum[r] = 0.0f; }

  const float scale = 0.08838834764831845f;  // 1/sqrt(128)
  int nchunks = (q0 + 16 + 31) >> 5;         // keys 0 .. q0+15, chunks of 32

  for (int c = 0; c < nchunks; ++c) {
    int k0 = c << 5;

    // --- S = Q * K^T (16 x 32 keys) ---
    v8f s0 = {}, s1 = {};
#pragma unroll
    for (int d = 0; d < 4; ++d) {
      v16bf kb0 = load_frag_b(Kb, QKV_COLS, k0, d * 32);
      v16bf kb1 = load_frag_b(Kb, QKV_COLS, k0 + 16, d * 32);
      s0 = wmma_bf16(qf[d], kb0, s0);
      s1 = wmma_bf16(qf[d], kb1, s1);
    }

    // --- scale + causal mask + online softmax ---
#pragma unroll
    for (int r = 0; r < 8; ++r) {
      int m = q0 + r + half * 8;
      float a = s0[r] * scale;
      float bsv = s1[r] * scale;
      if (k0 + ncol > m)      a   = -1e30f;
      if (k0 + 16 + ncol > m) bsv = -1e30f;
      float mx = fmaxf(a, bsv);
#pragma unroll
      for (int off = 1; off < 16; off <<= 1)
        mx = fmaxf(mx, __shfl_xor(mx, off, 32));
      float mnew = fmaxf(rmax[r], mx);
      float corr = __expf(rmax[r] - mnew);
      rmax[r] = mnew;
      float p0 = __expf(a - mnew);
      float p1 = __expf(bsv - mnew);
      float ls = p0 + p1;
#pragma unroll
      for (int off = 1; off < 16; off <<= 1)
        ls += __shfl_xor(ls, off, 32);
      rsum[r] = rsum[r] * corr + ls;
#pragma unroll
      for (int dt = 0; dt < 8; ++dt) o[dt][r] *= corr;
      s0[r] = p0;
      s1[r] = p1;
    }

    // --- stage P through LDS: C-layout -> A-layout fragment ---
    bf16_t* sp = &sP[wave][0][0];
#pragma unroll
    for (int r = 0; r < 8; ++r) {
      sp[(r + half * 8) * 32 + ncol]      = (bf16_t)s0[r];
      sp[(r + half * 8) * 32 + 16 + ncol] = (bf16_t)s1[r];
    }
    asm volatile("s_wait_dscnt 0x0" ::: "memory");
    const bf16_t* ap = sp + (lane & 15) * 32 + half * 8;
    v8bf plo = *(const v8bf*)ap;
    v8bf phi = *(const v8bf*)(ap + 16);
    v16bf pf;
#pragma unroll
    for (int i = 0; i < 8; ++i) { pf[i] = plo[i]; pf[8 + i] = phi[i]; }
    asm volatile("" ::: "memory");

    // --- O += P * V  (V^T rows are contiguous along keys) ---
#pragma unroll
    for (int dt = 0; dt < 8; ++dt) {
      v16bf vb = load_frag_b(Vh, SEQ, dt * 16, k0);
      o[dt] = wmma_bf16(pf, vb, o[dt]);
    }
  }

  // --- normalize and write O (bf16) ---
#pragma unroll
  for (int r = 0; r < 8; ++r) {
    float inv = 1.0f / rsum[r];
    int m = q0 + r + half * 8;
    size_t rowoff = ((size_t)(b * SEQ + m)) * DIM + h * HEAD_DIM;
#pragma unroll
    for (int dt = 0; dt < 8; ++dt)
      obf[rowoff + dt * 16 + ncol] = (bf16_t)(o[dt][r] * inv);
  }
}

// ---------------------------------------------------------------------------
// Launch
// ---------------------------------------------------------------------------
extern "C" void kernel_launch(void* const* d_in, const int* in_sizes, int n_in,
                              void* d_out, int out_size, void* d_ws, size_t ws_size,
                              hipStream_t stream) {
  const float* x      = (const float*)d_in[0];   // [B, T, DIM]
  const float* w_qkv  = (const float*)d_in[1];   // [3*DIM, DIM]
  const float* w_proj = (const float*)d_in[2];   // [DIM, DIM]
  float* out = (float*)d_out;                    // [B, T, DIM]

  const size_t nX  = (size_t)ROWS * DIM;         // 16,777,216
  const size_t nWq = (size_t)QKV_COLS * DIM;     // 12,582,912
  const size_t nWp = (size_t)DIM * DIM;          //  4,194,304
  const size_t nQKV = (size_t)ROWS * QKV_COLS;   // 50,331,648

  char* p = (char*)d_ws;
  bf16_t* Xbf  = (bf16_t*)p;            p += nX  * sizeof(bf16_t);
  bf16_t* Wqb  = (bf16_t*)p;            p += nWq * sizeof(bf16_t);
  bf16_t* Wpb  = (bf16_t*)p;            p += nWp * sizeof(bf16_t);
  bf16_t* QKV  = (bf16_t*)p;            p += nQKV * sizeof(bf16_t);
  bf16_t* Vt   = (bf16_t*)p;            p += nX  * sizeof(bf16_t);
  bf16_t* Obf  = (bf16_t*)p;            p += nX  * sizeof(bf16_t);

  f32_to_bf16_k<<<4096, 256, 0, stream>>>(x,      Xbf, (int)nX);
  f32_to_bf16_k<<<4096, 256, 0, stream>>>(w_qkv,  Wqb, (int)nWq);
  f32_to_bf16_k<<<4096, 256, 0, stream>>>(w_proj, Wpb, (int)nWp);

  // qkv = x @ w_qkv^T : M=8192, N=6144, K=2048
  gemm_wmma_k<true><<<dim3(ROWS / 128, QKV_COLS / 64), 256, 0, stream>>>(
      Xbf, Wqb, QKV, ROWS, QKV_COLS, DIM);

  // RoPE on Q,K in place; V -> Vt[b,h,d,t]
  rope_reorg_k<<<ROWS, 256, 0, stream>>>(QKV, Vt);

  // causal flash attention
  attention_k<<<dim3(SEQ / 128, BATCH * N_HEADS), 256, 0, stream>>>(QKV, Vt, Obf);

  // out = O @ w_proj^T : M=8192, N=2048, K=2048 (fp32 output)
  gemm_wmma_k<false><<<dim3(ROWS / 128, DIM / 64), 256, 0, stream>>>(
      Obf, Wpb, out, ROWS, DIM, DIM);
}
